// RepLKMatchingAdapter_84679575208383
// MI455X (gfx1250) — compile-verified, matrix operand order checked by reading, not run on hardware
//
#include <hip/hip_runtime.h>
#include <hip/hip_bf16.h>

#define B_ 2
#define F_ 2
#define C_ 128
#define D_ 96
#define H_ 48
#define W_ 128
#define MIND 0.1f
#define MAXD 20.0f
#define EPSF 1e-7f

typedef _Float16 v8h  __attribute__((ext_vector_type(8)));
typedef _Float16 v16h __attribute__((ext_vector_type(16)));
typedef float    v8f  __attribute__((ext_vector_type(8)));
typedef short    v8s  __attribute__((ext_vector_type(8)));

// |x| for packed f16 via sign-bit clear (lowers to 32-bit v_and on pairs)
__device__ __forceinline__ v8h habs8(v8h x) {
    v8s u;
    __builtin_memcpy(&u, &x, 16);
    u &= (short)0x7FFF;
    v8h r;
    __builtin_memcpy(&r, &u, 16);
    return r;
}

// ---------------------------------------------------------------------------
// Kernel 1: [N,C,H,W] f32 -> [N,H,W,C] f16 (channel-last: corner gathers become
// contiguous 256B runs, halves L2 traffic, feeds WMMA directly).
// ---------------------------------------------------------------------------
__global__ void cvt_chlast(const float* __restrict__ src, _Float16* __restrict__ dst, int total) {
    int idx = blockIdx.x * 256 + threadIdx.x;
    if (idx >= total) return;
    int c = idx % C_;
    int p = idx / C_;
    int w = p % W_;
    int t = p / W_;
    int h = t % H_;
    int n = t / H_;
    dst[idx] = (_Float16)src[(((size_t)n * C_ + c) * H_ + h) * W_ + w];
}

// ---------------------------------------------------------------------------
// Kernel 2: cost volume. One wave = 16 pixels of one (b,d,h) row-tile.
// Bilinear gather in packed f16; channel reduction via 4x chained
// v_wmma_f32_16x16x32_f16 against an all-ones B matrix (D[M,N] = row-sum S_M).
// Edge mask folded into corner weights + cur scaling: e*|g-cur| = |e*g - e*cur|.
// ---------------------------------------------------------------------------
__global__ __launch_bounds__(256) void costvol(
    const _Float16* __restrict__ curh,   // [B,H,W,C] f16
    const _Float16* __restrict__ lookh,  // [B,F,H,W,C] f16
    const float* __restrict__ poses,     // [B,F,4,4]
    const float* __restrict__ Kmat,      // [B,4,4]
    const float* __restrict__ invK,      // [B,4,4]
    float* __restrict__ cost_raw)        // [B,D,H,W] (d_out cost half)
{
    const int lane = threadIdx.x & 31;
    const int wave = threadIdx.x >> 5;
    const int m    = lane & 15;   // pixel within tile (A-matrix row M)
    const int hv   = lane >> 4;   // lane half selects K-chunk set

    int gt = blockIdx.x * 8 + wave;       // tile id over B*D*H*(W/16) = 73728
    int tw = gt & 7;  int t1 = gt >> 3;
    int h  = t1 % H_; int t2 = t1 / H_;
    int d  = t2 % D_; int b  = t2 / D_;
    int w  = tw * 16 + m;

    float depth = MIND + (float)d * ((MAXD - MIND) / (float)(D_ - 1));

    // Preload this lane's cur chunks (depth/frame invariant).
    const _Float16* curp = curh + (((size_t)(b * H_ + h)) * W_ + w) * C_;
    v8h curLo[4], curHi[4];
#pragma unroll
    for (int j = 0; j < 4; ++j) {
        int c0 = 32 * j + 8 * hv;
        curLo[j] = *(const v8h*)(curp + c0);
        curHi[j] = *(const v8h*)(curp + c0 + 16);
    }

    // Camera ray (align_corners=True collapses grid math to raw pixel coords).
    const float* iK = invK + b * 16;
    float px = (float)w, py = (float)h;
    float dirx = iK[0] * px + iK[1] * py + iK[2];
    float diry = iK[4] * px + iK[5] * py + iK[6];
    float dirz = iK[8] * px + iK[9] * py + iK[10];

    float border = (h >= 2 && h <= H_ - 3 && w >= 2 && w <= W_ - 3) ? 1.f : 0.f;

    v16h ones;
#pragma unroll
    for (int i = 0; i < 16; ++i) ones[i] = (_Float16)1.0f;

    v8f costAcc, cntAcc;
#pragma unroll
    for (int i = 0; i < 8; ++i) { costAcc[i] = 0.f; cntAcc[i] = 0.f; }

#pragma unroll
    for (int f = 0; f < F_; ++f) {
        const float* ps = poses + (size_t)(b * F_ + f) * 16;
        float psum = 0.f;
#pragma unroll
        for (int i = 0; i < 16; ++i) psum += ps[i];
        float validf = (psum != 0.f) ? 1.f : 0.f;

        // P = (K @ pose)[:3,:]
        const float* Kb = Kmat + b * 16;
        float P[3][4];
#pragma unroll
        for (int i = 0; i < 3; ++i)
#pragma unroll
            for (int k = 0; k < 4; ++k) {
                float s = 0.f;
#pragma unroll
                for (int t = 0; t < 4; ++t) s += Kb[i * 4 + t] * ps[t * 4 + k];
                P[i][k] = s;
            }

        float camx = depth * (P[0][0] * dirx + P[0][1] * diry + P[0][2] * dirz) + P[0][3];
        float camy = depth * (P[1][0] * dirx + P[1][1] * diry + P[1][2] * dirz) + P[1][3];
        float camz = depth * (P[2][0] * dirx + P[2][1] * diry + P[2][2] * dirz) + P[2][3];
        float inz = 1.0f / (camz + EPSF);
        float xs = camx * inz, ys = camy * inz;

        float x0f = floorf(xs), y0f = floorf(ys);
        float wx1 = xs - x0f, wx0 = 1.f - wx1;
        float wy1 = ys - y0f, wy0 = 1.f - wy1;

        float xcf[2] = {x0f, x0f + 1.f};
        float ycf[2] = {y0f, y0f + 1.f};
        float wxx[2] = {wx0, wx1};
        float wyy[2] = {wy0, wy1};

        float edge = ((xs >= 2.f) && (xs <= (float)(W_ - 2)) &&
                      (ys >= 2.f) && (ys <= (float)(H_ - 2))) ? border : 0.f;

        float cw[4]; int cx[4], cy[4];
#pragma unroll
        for (int cyi = 0; cyi < 2; ++cyi)
#pragma unroll
            for (int cxi = 0; cxi < 2; ++cxi) {
                int ci = cyi * 2 + cxi;
                float vflag = (xcf[cxi] >= 0.f && xcf[cxi] <= (float)(W_ - 1) &&
                               ycf[cyi] >= 0.f && ycf[cyi] <= (float)(H_ - 1)) ? 1.f : 0.f;
                // fold zero-padding validity AND edge mask into the weight (all 0/1 exact)
                cw[ci] = wyy[cyi] * wxx[cxi] * vflag * edge;
                cx[ci] = (int)fminf(fmaxf(xcf[cxi], 0.f), (float)(W_ - 1));
                cy[ci] = (int)fminf(fmaxf(ycf[cyi], 0.f), (float)(H_ - 1));
            }

        _Float16 he  = (_Float16)edge;
        _Float16 w0h = (_Float16)cw[0], w1h = (_Float16)cw[1];
        _Float16 w2h = (_Float16)cw[2], w3h = (_Float16)cw[3];

        const _Float16* lbase = lookh + ((size_t)(b * F_ + f)) * H_ * W_ * C_;
        const _Float16* cp0 = lbase + ((size_t)cy[0] * W_ + cx[0]) * C_;
        const _Float16* cp1 = lbase + ((size_t)cy[1] * W_ + cx[1]) * C_;
        const _Float16* cp2 = lbase + ((size_t)cy[2] * W_ + cx[2]) * C_;
        const _Float16* cp3 = lbase + ((size_t)cy[3] * W_ + cx[3]) * C_;

        v8f acc;
#pragma unroll
        for (int i = 0; i < 8; ++i) acc[i] = 0.f;

#pragma unroll
        for (int j = 0; j < 4; ++j) {
            int c0 = 32 * j + 8 * hv;   // lane-half-dependent K chunks (ISA A-layout)
            int c1 = c0 + 16;

            // warped * edge  (edge pre-folded into weights)
            v8h gLo = w0h * (*(const v8h*)(cp0 + c0)) + w1h * (*(const v8h*)(cp1 + c0))
                    + w2h * (*(const v8h*)(cp2 + c0)) + w3h * (*(const v8h*)(cp3 + c0));
            v8h gHi = w0h * (*(const v8h*)(cp0 + c1)) + w1h * (*(const v8h*)(cp1 + c1))
                    + w2h * (*(const v8h*)(cp2 + c1)) + w3h * (*(const v8h*)(cp3 + c1));

            // |e*g - e*cur| == e*|g - cur|  (e in {0,1})
            v8h aLo = habs8(gLo - he * curLo[j]);
            v8h aHi = habs8(gHi - he * curHi[j]);

            v16h A;
#pragma unroll
            for (int i = 0; i < 8; ++i) { A[i] = aLo[i]; A[i + 8] = aHi[i]; }

            // D[M,N] += sum_K A[M,K]  (all-ones B => every column holds the row sum)
            acc = __builtin_amdgcn_wmma_f32_16x16x32_f16(
                false, A, false, ones, (short)0, acc, false, false);
        }

        float scale = validf * (1.0f / (float)C_);
#pragma unroll
        for (int r = 0; r < 8; ++r) {
            float dv = acc[r] * scale;   // diffs for pixel (8*hv + r)
            costAcc[r] += dv;
            cntAcc[r]  += (dv > 0.f) ? 1.f : 0.f;
        }
    }

    // lanes 0 and 16 each own 8 consecutive pixel sums (C/D layout: VGPR r = M = r + 8*hv)
    if (m == 0) {
        size_t base = (((size_t)(b * D_ + d)) * H_ + h) * W_ + tw * 16 + hv * 8;
#pragma unroll
        for (int r = 0; r < 8; ++r)
            cost_raw[base + r] = costAcc[r] / (cntAcc[r] + EPSF);
    }
}

// ---------------------------------------------------------------------------
// Kernel 3: per (b,h,w): max over D, missing fill. In-place on the cost half.
// ---------------------------------------------------------------------------
__global__ void finalize(float* __restrict__ out) {
    int idx = blockIdx.x * 256 + threadIdx.x;  // B*H*W = 12288
    int w = idx % W_;
    int t = idx / W_;
    int h = t % H_;
    int b = t / H_;
    const size_t stride = (size_t)H_ * W_;
    float* oc = out;
    float* om = out + (size_t)B_ * D_ * stride;
    size_t base = ((size_t)b * D_) * stride + (size_t)h * W_ + w;

    float mx = -3.402823466e+38f;
    for (int d = 0; d < D_; ++d) mx = fmaxf(mx, oc[base + (size_t)d * stride]);
    for (int d = 0; d < D_; ++d) {
        size_t o = base + (size_t)d * stride;
        float v = oc[o];
        float miss = (v == 0.f) ? 1.f : 0.f;
        oc[o] = v * (1.f - miss) + mx * miss;
        om[o] = miss;
    }
}

extern "C" void kernel_launch(void* const* d_in, const int* in_sizes, int n_in,
                              void* d_out, int out_size, void* d_ws, size_t ws_size,
                              hipStream_t stream) {
    (void)in_sizes; (void)n_in; (void)out_size; (void)ws_size;
    const float* cur   = (const float*)d_in[0];  // [B,C,H,W]
    const float* look  = (const float*)d_in[1];  // [B,F,C,H,W]
    const float* poses = (const float*)d_in[2];  // [B,F,4,4]
    const float* Km    = (const float*)d_in[3];  // [B,4,4]
    const float* iK    = (const float*)d_in[4];  // [B,4,4]

    const size_t curBytes = (size_t)B_ * H_ * W_ * C_ * sizeof(_Float16);        // 3 MB
    _Float16* curh  = (_Float16*)d_ws;
    _Float16* lookh = (_Float16*)((char*)d_ws + curBytes);                        // 6 MB

    const int curTot  = B_ * H_ * W_ * C_;        // 1,572,864
    const int lookTot = B_ * F_ * H_ * W_ * C_;   // 3,145,728

    cvt_chlast<<<curTot / 256, 256, 0, stream>>>(cur, curh, curTot);
    cvt_chlast<<<lookTot / 256, 256, 0, stream>>>(look, lookh, lookTot);

    // 73728 wave-tiles, 8 waves (256 threads) per block
    costvol<<<(B_ * D_ * H_ * (W_ / 16)) / 8, 256, 0, stream>>>(
        curh, lookh, poses, Km, iK, (float*)d_out);

    finalize<<<(B_ * H_ * W_) / 256, 256, 0, stream>>>((float*)d_out);
}
